// SA_Memory_Module_WithAttention_48979807043761
// MI455X (gfx1250) — matrix operand order repaired and checked by reading, not compiled, not run on hardware
//
#include <hip/hip_runtime.h>
#include <hip/hip_bf16.h>
#include <cmath>

// ---------------------------------------------------------------------------
// SA memory module, MI455X (gfx1250, wave32, WMMA), LDS-staged bf16 GEMMs.
//
// Algebraic restructuring (exact up to fp reassociation):
//   * pooling commutes with conv1x1  -> project pooled [B,C,P] tensors (256^3 GEMMs)
//   * bilinear upsample commutes with conv1x1 over channels ->
//       Zc = wz @ [Zh;Zm] + bz                  (on 16x16 grid)
//       G  = wm[:, :512] @ Zc                   (on 16x16 grid)
//       combined(pixel) = interp(G) + wm[:,512:768] @ h(pixel) + bm
//   Only the last GEMM runs at full 128x128 resolution (fused with gating);
//   its weight slice is pre-converted once to a fragment-ready bf16 buffer.
// ---------------------------------------------------------------------------

typedef __attribute__((ext_vector_type(16))) __bf16 v16bf;
typedef __attribute__((ext_vector_type(8)))  float  v8f;

union BfVec { v16bf v; unsigned short u[16]; };

__device__ __forceinline__ unsigned short f2bf(float f) {
    unsigned int x = __float_as_uint(f);
    x += 0x7fffu + ((x >> 16) & 1u);   // round-to-nearest-even
    return (unsigned short)(x >> 16);
}

// WMMA fragment-ready slot maps (16x16x32 bf16):
//   A (16m x 32k): lane = ((k>>3)&1)*16 + (m&15); e = (k>>4)*8 + (k&7)
//   B (32k x 16n): lane = ((k>>3)&1)*16 + (n&15); e = (k>>4)*8 + (k&7)
__device__ __forceinline__ void putA(BfVec* tile, int mtile, int m, int k, unsigned short v) {
    int lane = (((k >> 3) & 1) << 4) | (m & 15);
    int e    = ((k >> 4) << 3) | (k & 7);
    tile[mtile * 32 + lane].u[e] = v;
}
__device__ __forceinline__ void putB(BfVec* tile, int ntile, int k, int n, unsigned short v) {
    int lane = (((k >> 3) & 1) << 4) | (n & 15);
    tile[ntile * 32 + lane].u[((k >> 4) << 3) | (k & 7)] = v;
}

// ---------------- constants --------------------------------------------------
#define BB 8
#define CC 256
#define HWW 16384              // 128*128
#define PP 256                 // patches
#define BCP (BB * CC * PP)     // 524288

// workspace offsets (in floats)
#define OFF_HP   ((size_t)0)
#define OFF_MP   (OFF_HP  + BCP)
#define OFF_QP   (OFF_MP  + BCP)
#define OFF_KP   (OFF_QP  + BCP)
#define OFF_VP   (OFF_KP  + BCP)
#define OFF_KMP  (OFF_VP  + BCP)
#define OFF_VMP  (OFF_KMP + BCP)
#define OFF_SH   (OFF_VMP + BCP)              // [B,P,P]
#define OFF_SM   (OFF_SH  + BCP)              // contiguous with SH
#define OFF_ZCAT (OFF_SM  + BCP)              // [B,512,P]
#define OFF_ZC   (OFF_ZCAT + (size_t)BB*512*PP)
#define OFF_G    (OFF_ZC   + (size_t)BB*512*PP)   // [B,768,P]
#define OFF_WF   (OFF_G    + (size_t)BB*768*PP)   // wm2 frag-ready bf16 (98304 floats)

// ---------------- 1) patch pooling ------------------------------------------
__global__ __launch_bounds__(256) void pool_kernel(
    const float* __restrict__ h, const float* __restrict__ m,
    float* __restrict__ Hp, float* __restrict__ Mp, int nchunks)
{
    int ci   = blockIdx.x * 8 + (threadIdx.x >> 5);
    int lane = threadIdx.x & 31;
    const float* src; float* dst; int c;
    if (ci < nchunks) { src = h; dst = Hp; c = ci; }
    else              { src = m; dst = Mp; c = ci - nchunks; }
    const float* p = src + (size_t)c * 64;
    float s = p[lane] + p[lane + 32];
    #pragma unroll
    for (int off = 16; off; off >>= 1) s += __shfl_xor(s, off);
    if (lane == 0) dst[c] = s * (1.0f / 64.0f);
}

// ---------------- 2) generic batched bf16 WMMA GEMM (LDS staged) ------------
// Block tile 128(M) x 64(N) x 32(K); 8 waves, each owns a 32x32 tile.
// a(m,k) = TRANSA ? A[k*lda+m] : A[m*lda+k]   (same for B).
template <bool TRANSA, bool TRANSB>
__global__ __launch_bounds__(256) void gemm_wmma_kernel(
    const float* __restrict__ A, unsigned long long sAb, int lda,
    const float* __restrict__ B, unsigned long long sBb, int ldb,
    float*       __restrict__ C, unsigned long long sCb, int ldc,
    int K, const float* __restrict__ bias)
{
    __shared__ BfVec lA[8 * 32];   // 8 KiB, 8 m-tiles
    __shared__ BfVec lB[4 * 32];   // 4 KiB, 4 n-tiles

    const float* Ab = A + sAb * blockIdx.z;
    const float* Bb = B + sBb * blockIdx.z;
    float*       Cb = C + sCb * blockIdx.z;
    const int t    = threadIdx.x;
    const int lane = t & 31;
    const int wv   = t >> 5;
    const int lo   = lane & 15;
    const int hi   = lane >> 4;
    const int mblk = blockIdx.y * 128;
    const int nblk = blockIdx.x * 64;
    const int wm_t = (wv & 3) * 2;   // wave m-tile base (of 8)
    const int wn_t = (wv >> 2) * 2;  // wave n-tile base (of 4)

    v8f acc[2][2] = {};
    for (int k0 = 0; k0 < K; k0 += 32) {
        // ---- stage A tile 128x32 (16 elems / thread, coalesced) ----
        if (!TRANSA) {
            int m = t >> 1, kb = (t & 1) * 16;
            const float* src = Ab + (size_t)(mblk + m) * lda + k0 + kb;
            #pragma unroll
            for (int e2 = 0; e2 < 16; ++e2)
                putA(lA, m >> 4, m, kb + e2, f2bf(src[e2]));
        } else {
            int k = t >> 3, mb = (t & 7) * 16;
            const float* src = Ab + (size_t)(k0 + k) * lda + mblk + mb;
            #pragma unroll
            for (int e2 = 0; e2 < 16; ++e2)
                putA(lA, (mb + e2) >> 4, mb + e2, k, f2bf(src[e2]));
        }
        // ---- stage B tile 32x64 (8 elems / thread, coalesced) ----
        if (!TRANSB) {
            int k = t >> 3, nb = (t & 7) * 8;
            const float* src = Bb + (size_t)(k0 + k) * ldb + nblk + nb;
            #pragma unroll
            for (int e2 = 0; e2 < 8; ++e2)
                putB(lB, (nb + e2) >> 4, k, nb + e2, f2bf(src[e2]));
        } else {
            int n = t >> 2, kb = (t & 3) * 8;
            const float* src = Bb + (size_t)(nblk + n) * ldb + k0 + kb;
            #pragma unroll
            for (int e2 = 0; e2 < 8; ++e2)
                putB(lB, n >> 4, kb + e2, n, f2bf(src[e2]));
        }
        __syncthreads();
        // ---- fragments straight from LDS (2x ds_load_b128 each) ----
        BfVec af[2], bf[2];
        #pragma unroll
        for (int mt = 0; mt < 2; ++mt) af[mt] = lA[(wm_t + mt) * 32 + lane];
        #pragma unroll
        for (int nt = 0; nt < 2; ++nt) bf[nt] = lB[(wn_t + nt) * 32 + lane];
        #pragma unroll
        for (int mt = 0; mt < 2; ++mt)
            #pragma unroll
            for (int nt = 0; nt < 2; ++nt)
                acc[mt][nt] = __builtin_amdgcn_wmma_f32_16x16x32_bf16(
                    false, af[mt].v, false, bf[nt].v,
                    (short)0, acc[mt][nt], false, false);
        __syncthreads();
    }
    #pragma unroll
    for (int mt = 0; mt < 2; ++mt)
        #pragma unroll
        for (int nt = 0; nt < 2; ++nt)
            #pragma unroll
            for (int i = 0; i < 8; ++i) {
                int m = mblk + (wm_t + mt) * 16 + (hi ? i + 8 : i);
                int n = nblk + (wn_t + nt) * 16 + lo;
                float v = acc[mt][nt][i];
                if (bias) v += bias[m];
                Cb[(size_t)m * ldc + n] = v;
            }
}

// ---------------- 3) softmax over rows of 256 -------------------------------
__global__ __launch_bounds__(256) void softmax_kernel(float* __restrict__ S)
{
    int row  = blockIdx.x * 8 + (threadIdx.x >> 5);
    int lane = threadIdx.x & 31;
    float* p = S + (size_t)row * PP;
    float v[8];
    #pragma unroll
    for (int j = 0; j < 8; ++j) v[j] = p[lane + j * 32];
    float mx = v[0];
    #pragma unroll
    for (int j = 1; j < 8; ++j) mx = fmaxf(mx, v[j]);
    #pragma unroll
    for (int off = 16; off; off >>= 1) mx = fmaxf(mx, __shfl_xor(mx, off));
    float sum = 0.0f;
    #pragma unroll
    for (int j = 0; j < 8; ++j) { v[j] = __expf(v[j] - mx); sum += v[j]; }
    #pragma unroll
    for (int off = 16; off; off >>= 1) sum += __shfl_xor(sum, off);
    float inv = 1.0f / sum;
    #pragma unroll
    for (int j = 0; j < 8; ++j) p[lane + j * 32] = v[j] * inv;
}

// ---------------- 4a) wm2 -> fragment-ready bf16 (once per call) ------------
// Record (ks, T, lane) -> 16 bf16 (32 B): launch-invariant, L2-resident after.
__global__ __launch_bounds__(256) void prep_wm_kernel(
    const float* __restrict__ wm, unsigned short* __restrict__ Wf)
{
    int rec  = blockIdx.x * 256 + threadIdx.x;   // 8*48*32 = 12288 records
    int lane = rec & 31;
    int T    = (rec >> 5) % 48;                  // m-tile (row r = T*16 + lo)
    int ks   = rec / (48 * 32);                  // k-step
    int hi   = (lane >> 4) & 1;
    const float* src = wm + (size_t)(T * 16 + (lane & 15)) * 768 + 512 + ks * 32;
    unsigned short* dst = Wf + (size_t)rec * 16;
    #pragma unroll
    for (int e = 0; e < 16; ++e) {
        int k = ((e >> 3) << 4) | (hi << 3) | (e & 7);
        dst[e] = f2bf(src[k]);
    }
}

// ---------------- 4b) fused final GEMM + interp(G) + gating -----------------
// Per workgroup: 32 pixels x all 768 combined channels; h tile via LDS,
// wm2 fragments read directly (coalesced 32B/lane) from the prepped buffer.
// wave w owns rows {w*32..w*32+31} + {0,256,512} -> mo/mg/mi co-resident.
__global__ __launch_bounds__(256) void fuse_out_kernel(
    const float* __restrict__ h, const float* __restrict__ m,
    const BfVec* __restrict__ Wf, const float* __restrict__ bm,
    const float* __restrict__ G,
    float* __restrict__ outh, float* __restrict__ outm)
{
    __shared__ BfVec lH[2 * 32];   // 2 KiB: h tile, 32 k x 32 pixels

    const int t     = threadIdx.x;
    const int lane  = t & 31;
    const int wv    = t >> 5;
    const int lo    = lane & 15;
    const int hi    = lane >> 4;
    const int b     = blockIdx.y;
    const int pbase = blockIdx.x * 32;

    v8f acc[3][2][2] = {};   // [mo|mg|mi][row sub-tile][pixel sub-tile]
    for (int ks = 0; ks < 8; ++ks) {
        // stage h tile: 32k x 32pix (4 elems / thread)
        {
            int k = t >> 3, nb = (t & 7) * 4;
            const float* src = h + ((size_t)(b * CC + ks * 32 + k)) * HWW + pbase + nb;
            #pragma unroll
            for (int e2 = 0; e2 < 4; ++e2)
                putB(lH, (nb + e2) >> 4, k, nb + e2, f2bf(src[e2]));
        }
        __syncthreads();
        BfVec bf[2];
        #pragma unroll
        for (int nt = 0; nt < 2; ++nt) bf[nt] = lH[nt * 32 + lane];
        #pragma unroll
        for (int g = 0; g < 3; ++g)
            #pragma unroll
            for (int s = 0; s < 2; ++s) {
                BfVec af = Wf[((size_t)ks * 48 + (g * 16 + wv * 2 + s)) * 32 + lane];
                #pragma unroll
                for (int nt = 0; nt < 2; ++nt)
                    acc[g][s][nt] = __builtin_amdgcn_wmma_f32_16x16x32_bf16(
                        false, af.v, false, bf[nt].v,
                        (short)0, acc[g][s][nt], false, false);
            }
        __syncthreads();
    }

    const float* Gb = G + (size_t)b * 768 * PP;
    #pragma unroll
    for (int nt = 0; nt < 2; ++nt) {
        int pix = pbase + nt * 16 + lo;
        int y = pix >> 7, x = pix & 127;
        // half-pixel bilinear 16x16 -> 128x128 (matches jax.image.resize)
        float fy = (y + 0.5f) * 0.125f - 0.5f;
        float fx = (x + 0.5f) * 0.125f - 0.5f;
        int y0 = (int)floorf(fy), x0 = (int)floorf(fx);
        float ty = fy - (float)y0, tx = fx - (float)x0;
        int y1 = min(max(y0 + 1, 0), 15); y0 = min(max(y0, 0), 15);
        int x1 = min(max(x0 + 1, 0), 15); x0 = min(max(x0, 0), 15);
        float w00 = (1.f - ty) * (1.f - tx), w01 = (1.f - ty) * tx;
        float w10 = ty * (1.f - tx),         w11 = ty * tx;
        int p00 = y0 * 16 + x0, p01 = y0 * 16 + x1;
        int p10 = y1 * 16 + x0, p11 = y1 * 16 + x1;
        #pragma unroll
        for (int s = 0; s < 2; ++s)
            #pragma unroll
            for (int i = 0; i < 8; ++i) {
                int r = wv * 32 + s * 16 + (hi ? i + 8 : i);
                const float* g0 = Gb + (size_t)r * PP;
                const float* g1 = Gb + (size_t)(256 + r) * PP;
                const float* g2 = Gb + (size_t)(512 + r) * PP;
                float mo = acc[0][s][nt][i] + bm[r] +
                    w00 * g0[p00] + w01 * g0[p01] + w10 * g0[p10] + w11 * g0[p11];
                float mg = acc[1][s][nt][i] + bm[256 + r] +
                    w00 * g1[p00] + w01 * g1[p01] + w10 * g1[p10] + w11 * g1[p11];
                float mi = acc[2][s][nt][i] + bm[512 + r] +
                    w00 * g2[p00] + w01 * g2[p01] + w10 * g2[p10] + w11 * g2[p11];
                size_t idx = ((size_t)(b * CC + r)) * HWW + pix;
                float mv = m[idx];
                float gi = 1.0f / (1.0f + __expf(-mi));
                float nm = (1.0f - gi) * mv + gi * tanhf(mg);
                float nh = nm / (1.0f + __expf(-mo));
                outh[idx] = nh;
                outm[idx] = nm;
            }
    }
}

// ---------------------------------------------------------------------------
extern "C" void kernel_launch(void* const* d_in, const int* in_sizes, int n_in,
                              void* d_out, int out_size, void* d_ws, size_t ws_size,
                              hipStream_t stream) {
    const float* h   = (const float*)d_in[0];
    const float* m   = (const float*)d_in[1];
    const float* wq  = (const float*)d_in[2];
    const float* bq  = (const float*)d_in[3];
    const float* wk  = (const float*)d_in[4];
    const float* bk  = (const float*)d_in[5];
    const float* wv  = (const float*)d_in[6];
    const float* bv  = (const float*)d_in[7];
    const float* wkm = (const float*)d_in[8];
    const float* bkm = (const float*)d_in[9];
    const float* wvm = (const float*)d_in[10];
    const float* bvm = (const float*)d_in[11];
    const float* wz  = (const float*)d_in[12];
    const float* bz  = (const float*)d_in[13];
    const float* wmw = (const float*)d_in[14];
    const float* bm  = (const float*)d_in[15];

    float* ws = (float*)d_ws;
    float* Hp   = ws + OFF_HP;
    float* Mp   = ws + OFF_MP;
    float* Qp   = ws + OFF_QP;
    float* Kp   = ws + OFF_KP;
    float* Vp   = ws + OFF_VP;
    float* Kmp  = ws + OFF_KMP;
    float* Vmp  = ws + OFF_VMP;
    float* Sh   = ws + OFF_SH;
    float* Sm   = ws + OFF_SM;
    float* Zcat = ws + OFF_ZCAT;
    float* Zc   = ws + OFF_ZC;
    float* G    = ws + OFF_G;
    unsigned short* Wf = (unsigned short*)(ws + OFF_WF);

    const unsigned long long sDP = (unsigned long long)CC * PP;   // 65536
    const unsigned long long s2D = (unsigned long long)512 * PP;  // 131072
    const unsigned long long s3D = (unsigned long long)768 * PP;  // 196608

    // 1) pool h and m -> [B,C,P];  prep wm2 fragments (independent)
    pool_kernel<<<(2 * BCP) / 8, 256, 0, stream>>>(h, m, Hp, Mp, BCP);
    prep_wm_kernel<<<48, 256, 0, stream>>>(wmw, Wf);

    // 2) projections (pool/conv commuted): per-batch 256x256x256 GEMMs
    dim3 g256(4, 2, BB);
    gemm_wmma_kernel<false,false><<<g256, 256, 0, stream>>>(wq,  0, CC, Hp, sDP, PP, Qp,  sDP, PP, CC, bq);
    gemm_wmma_kernel<false,false><<<g256, 256, 0, stream>>>(wk,  0, CC, Hp, sDP, PP, Kp,  sDP, PP, CC, bk);
    gemm_wmma_kernel<false,false><<<g256, 256, 0, stream>>>(wv,  0, CC, Hp, sDP, PP, Vp,  sDP, PP, CC, bv);
    gemm_wmma_kernel<false,false><<<g256, 256, 0, stream>>>(wkm, 0, CC, Mp, sDP, PP, Kmp, sDP, PP, CC, bkm);
    gemm_wmma_kernel<false,false><<<g256, 256, 0, stream>>>(wvm, 0, CC, Mp, sDP, PP, Vmp, sDP, PP, CC, bvm);

    // 3) attention scores  S = Qp^T @ K  -> softmax (Sh,Sm contiguous)
    gemm_wmma_kernel<true,false><<<g256, 256, 0, stream>>>(Qp, sDP, PP, Kp,  sDP, PP, Sh, sDP, PP, CC, nullptr);
    gemm_wmma_kernel<true,false><<<g256, 256, 0, stream>>>(Qp, sDP, PP, Kmp, sDP, PP, Sm, sDP, PP, CC, nullptr);
    softmax_kernel<<<(2 * BB * PP) / 8, 256, 0, stream>>>(Sh);

    // 4) Z_img[d,p] = V @ A^T   (Zcat halves: Zh rows 0..255, Zm 256..511)
    gemm_wmma_kernel<false,true><<<g256, 256, 0, stream>>>(Vp,  sDP, PP, Sh, sDP, PP, Zcat,           s2D, PP, PP, nullptr);
    gemm_wmma_kernel<false,true><<<g256, 256, 0, stream>>>(Vmp, sDP, PP, Sm, sDP, PP, Zcat + CC * PP, s2D, PP, PP, nullptr);

    // 5) fold the big conv1x1s onto the 16x16 grid (interp commutes with conv)
    gemm_wmma_kernel<false,false><<<dim3(4, 4, BB), 256, 0, stream>>>(wz,  0, 512, Zcat, s2D, PP, Zc, s2D, PP, 512, bz);
    gemm_wmma_kernel<false,false><<<dim3(4, 6, BB), 256, 0, stream>>>(wmw, 0, 768, Zc,   s2D, PP, G,  s3D, PP, 512, nullptr);

    // 6) fused full-res GEMM (wm[:,512:768] @ h) + interp(G) + bm + gating
    float* outh = (float*)d_out;
    float* outm = outh + (size_t)BB * CC * HWW;
    fuse_out_kernel<<<dim3(HWW / 32, BB), 256, 0, stream>>>(
        h, m, (const BfVec*)Wf, bm, G, outh, outm);
}
